// PKSpellHierarchical_app1_1176821039633
// MI455X (gfx1250) — compile-verified, matrix operand order checked by reading, not run on hardware
//
#include <hip/hip_runtime.h>
#include <hip/hip_bf16.h>

// ---------------------------------------------------------------------------
// PKSpell hierarchical bi-GRU + attention + CE, CDNA5 (gfx1250) WMMA version.
// All matmuls (input gates, recurrent gates, q/v projections, attention
// scores, output heads) use v_wmma_f32_16x16x32_f16 (f16 in, f32 acc).
// ---------------------------------------------------------------------------

typedef _Float16 v16h __attribute__((ext_vector_type(16)));
typedef _Float16 v8h  __attribute__((ext_vector_type(8)));
typedef float    v8f  __attribute__((ext_vector_type(8)));

#define T_LEN   1024
#define BATCH   64
#define D_IN    17
#define HM      150     // main GRU hidden per dir
#define GM      450     // 3*HM
#define HMP     160     // HM padded to k-tile multiple
#define HHI     128     // hier GRU hidden per dir
#define GHI     384     // 3*HHI
#define NSEQ    4096    // B * N_MEAS
#define ROWS_TB 65536   // T*B
#define NPITCH  35
#define NKS     15
#define PAD_P   34
#define PAD_K   14

// ---- WMMA helpers ----------------------------------------------------------

// Load an A-fragment (16x32 f16, M x K) from a row-major f16 matrix.
// For f16 WMMA this same loader, applied to row-major W[N][K], produces the
// B-fragment of B = W^T (lane -> column N, elems -> K), so it serves both.
__device__ __forceinline__ v16h load_frag_rm(const _Float16* __restrict__ base,
                                             int ld, int r0, int k0) {
  const int l   = threadIdx.x & 31;
  const int row = r0 + (l & 15);
  const int kb  = k0 + ((l >> 4) << 3);     // high half-wave holds K+8..15 / K+24..31
  const _Float16* p = base + (size_t)row * ld + kb;
  v8h lo = *(const v8h*)(p);
  v8h hi = *(const v8h*)(p + 16);
  return __builtin_shufflevector(lo, hi, 0,1,2,3,4,5,6,7,8,9,10,11,12,13,14,15);
}

__device__ __forceinline__ v8f wmma16(v16h a, v16h b, v8f c) {
  return __builtin_amdgcn_wmma_f32_16x16x32_f16(false, a, false, b,
                                                (short)0, c, false, false);
}

// Store a 16x16 f32 accumulator tile (+bias) to f32 and/or f16 destinations.
__device__ __forceinline__ void store_tile(v8f acc,
                                           float* Cf, int ldf,
                                           _Float16* Ch, int ldh,
                                           int r0, int c0, int ncols,
                                           const float* bias) {
  const int l   = threadIdx.x & 31;
  const int col = c0 + (l & 15);
  if (col >= ncols) return;
  const float bv   = bias ? bias[col] : 0.f;
  const int  rbase = r0 + ((l >> 4) << 3);
#pragma unroll
  for (int i = 0; i < 8; ++i) {
    const float v = acc[i] + bv;
    const int row = rbase + i;
    if (Cf) Cf[(size_t)row * ldf + col] = v;
    if (Ch) Ch[(size_t)row * ldh + col] = (_Float16)v;
  }
}

__device__ __forceinline__ float sigm(float x) { return 1.f / (1.f + expf(-x)); }

// ---- generic WMMA GEMM: C[M,N] = A[M,Kpad] * W[N,Kpad]^T (+bias) ----------
// block = 128 threads (4 waves); grid = (M/64, Ntiles); each wave: 16x16 tile.
__global__ void gemm_wmma(const _Float16* __restrict__ A, int lda,
                          const _Float16* __restrict__ Bt, int ldb,
                          const float* __restrict__ bias,
                          float* __restrict__ Cf, int ldf,
                          _Float16* __restrict__ Ch, int ldh,
                          int ncols, int ktiles) {
  const int wid = threadIdx.x >> 5;
  const int r0  = blockIdx.x * 64 + wid * 16;
  const int c0  = blockIdx.y * 16;
  v8f acc = {};
  for (int kt = 0; kt < ktiles; ++kt) {
    if (kt + 1 < ktiles)  // exercise global_prefetch_b8 on the weight stream
      __builtin_prefetch(Bt + (size_t)c0 * ldb + (kt + 1) * 32, 0, 1);
    v16h a = load_frag_rm(A,  lda, r0, kt * 32);
    v16h b = load_frag_rm(Bt, ldb, c0, kt * 32);
    acc = wmma16(a, b, acc);
  }
  store_tile(acc, Cf, ldf, Ch, ldh, r0, c0, ncols, bias);
}

// ---- pack: f32 [N,K] -> f16 [Npad,Kpad] (zero padded) ----------------------
__global__ void pack_w16(const float* __restrict__ W, _Float16* __restrict__ out,
                         int N, int K, int Npad, int Kpad) {
  const int i = blockIdx.x * blockDim.x + threadIdx.x;
  if (i >= Npad * Kpad) return;
  const int r = i / Kpad, k = i - r * Kpad;
  out[i] = (r < N && k < K) ? (_Float16)W[(size_t)r * K + k] : (_Float16)0.f;
}

__global__ void zero_h16(_Float16* __restrict__ p, size_t n) {
  for (size_t i = blockIdx.x * (size_t)blockDim.x + threadIdx.x; i < n;
       i += (size_t)gridDim.x * blockDim.x)
    p[i] = (_Float16)0.f;
}

__global__ void zero_f32(float* __restrict__ p, int n) {
  const int i = blockIdx.x * blockDim.x + threadIdx.x;
  if (i < n) p[i] = 0.f;
}

// ---- main GRU persistent scan ---------------------------------------------
// grid = 4 blocks: (dir, batch-half). 256 threads = 8 waves. h kept in LDS.
__global__ void main_scan(const _Float16* __restrict__ gi_f,
                          const _Float16* __restrict__ gi_b,
                          const _Float16* __restrict__ whh_f,   // f16 [464,160]
                          const _Float16* __restrict__ whh_b,
                          const float* __restrict__ bhh_f,
                          const float* __restrict__ bhh_b,
                          _Float16* __restrict__ hx,            // [65536,320]
                          _Float16* __restrict__ stacked) {     // [65536,576]
  extern __shared__ char smem[];
  _Float16* h16 = (_Float16*)smem;                              // 32 x 160
  float*    hf  = (float*)(smem + 32 * HMP * 2);                // 32 x 150
  float*    gh  = (float*)(smem + 32 * HMP * 2 + 32 * HM * 4);  // 32 x 464
  const int dir  = blockIdx.x >> 1;
  const int half = blockIdx.x & 1;
  const int tid  = threadIdx.x;
  const int wid  = tid >> 5;
  const int nw   = blockDim.x >> 5;
  const _Float16* gi  = dir ? gi_b  : gi_f;
  const _Float16* whh = dir ? whh_b : whh_f;
  const float*    bhh = dir ? bhh_b : bhh_f;

  for (int i = tid; i < 32 * HMP; i += blockDim.x) h16[i] = (_Float16)0.f;
  for (int i = tid; i < 32 * HM;  i += blockDim.x) hf[i]  = 0.f;
  __syncthreads();

  for (int step = 0; step < T_LEN; ++step) {
    const int t = dir ? (T_LEN - 1 - step) : step;
    // gh = h @ Whh^T + bhh  (2 row-tiles x 29 col-tiles, 5 k-tiles)
    for (int tile = wid; tile < 2 * 29; tile += nw) {
      const int rt = tile & 1;
      const int ct = tile >> 1;
      v8f acc = {};
#pragma unroll
      for (int kt = 0; kt < 5; ++kt) {
        v16h a = load_frag_rm(h16, HMP, rt * 16, kt * 32);
        v16h b = load_frag_rm(whh, HMP, ct * 16, kt * 32);
        acc = wmma16(a, b, acc);
      }
      store_tile(acc, gh, 464, nullptr, 0, rt * 16, ct * 16, GM, bhh);
    }
    __syncthreads();
    // elementwise gate update (torch gate order r,z,n)
    for (int i = tid; i < 32 * HM; i += blockDim.x) {
      const int bl = i / HM;
      const int c  = i - bl * HM;
      const int bg = half * 32 + bl;
      const size_t gr = ((size_t)t * BATCH + bg) * GM;
      const float ir = (float)gi[gr + c];
      const float iz = (float)gi[gr + HM + c];
      const float in = (float)gi[gr + 2 * HM + c];
      const int gb = bl * 464;
      const float r = sigm(ir + gh[gb + c]);
      const float z = sigm(iz + gh[gb + HM + c]);
      const float n = tanhf(in + r * gh[gb + 2 * HM + c]);
      const float h = hf[bl * HM + c];
      const float hnew = (1.f - z) * n + z * h;
      hf[bl * HM + c]   = hnew;
      h16[bl * HMP + c] = (_Float16)hnew;
      const int meas = t >> 4, m = t & 15;
      hx[((size_t)m * NSEQ + bg * 64 + meas) * 320 + dir * HM + c] = (_Float16)hnew;
      stacked[((size_t)t * BATCH + bg) * 576 + dir * HM + c]       = (_Float16)hnew;
    }
    __syncthreads();
  }
}

// ---- hier GRU persistent scan ---------------------------------------------
// grid = 256 blocks: (group of 32 seqs) x dir. 256 threads = 8 waves.
__global__ void hier_scan(const _Float16* __restrict__ gih_f,
                          const _Float16* __restrict__ gih_b,
                          const _Float16* __restrict__ whh_f,   // f16 [384,128]
                          const _Float16* __restrict__ whh_b,
                          const float* __restrict__ bhh_f,
                          const float* __restrict__ bhh_b,
                          _Float16* __restrict__ hv) {          // [65536,256]
  extern __shared__ char smem[];
  _Float16* h16 = (_Float16*)smem;                               // 32 x 128
  float*    hf  = (float*)(smem + 32 * HHI * 2);                 // 32 x 128
  float*    gh  = (float*)(smem + 32 * HHI * 2 + 32 * HHI * 4);  // 32 x 384
  const int dir = blockIdx.x & 1;
  const int n0  = (blockIdx.x >> 1) * 32;
  const int tid = threadIdx.x;
  const int wid = tid >> 5;
  const int nw  = blockDim.x >> 5;
  const _Float16* gi  = dir ? gih_b : gih_f;
  const _Float16* whh = dir ? whh_b : whh_f;
  const float*    bhh = dir ? bhh_b : bhh_f;

  for (int i = tid; i < 32 * HHI; i += blockDim.x) { h16[i] = (_Float16)0.f; hf[i] = 0.f; }
  __syncthreads();

  for (int step = 0; step < 16; ++step) {
    const int m = dir ? (15 - step) : step;
    for (int tile = wid; tile < 2 * 24; tile += nw) {
      const int rt = tile & 1;
      const int ct = tile >> 1;
      v8f acc = {};
#pragma unroll
      for (int kt = 0; kt < 4; ++kt) {
        v16h a = load_frag_rm(h16, HHI, rt * 16, kt * 32);
        v16h b = load_frag_rm(whh, HHI, ct * 16, kt * 32);
        acc = wmma16(a, b, acc);
      }
      store_tile(acc, gh, GHI, nullptr, 0, rt * 16, ct * 16, GHI, bhh);
    }
    __syncthreads();
    for (int i = tid; i < 32 * HHI; i += blockDim.x) {
      const int bl = i >> 7;
      const int c  = i & 127;
      const int n  = n0 + bl;
      const size_t gr = ((size_t)m * NSEQ + n) * GHI;
      const float r = sigm((float)gi[gr + c]           + gh[bl * GHI + c]);
      const float z = sigm((float)gi[gr + HHI + c]     + gh[bl * GHI + HHI + c]);
      const float nn = tanhf((float)gi[gr + 2 * HHI + c] + r * gh[bl * GHI + 2 * HHI + c]);
      const float h = hf[bl * HHI + c];
      const float hnew = (1.f - z) * nn + z * h;
      hf[bl * HHI + c]  = hnew;
      h16[bl * HHI + c] = (_Float16)hnew;
      hv[((size_t)n * 16 + m) * 256 + dir * HHI + c] = (_Float16)hnew;
    }
    __syncthreads();
  }
}

// ---- attention: one wave per sequence n -----------------------------------
// S = q @ v^T via WMMA (B-frag of v^T == A-frag loader on row-major v),
// softmax over m, column-sum -> w, ctx = w @ hv.
__global__ void attention(const _Float16* __restrict__ q16,
                          const _Float16* __restrict__ v16,
                          const _Float16* __restrict__ hv,
                          _Float16* __restrict__ ctx16) {
  __shared__ float smem[8 * 288];
  const int wid = threadIdx.x >> 5;
  const int l   = threadIdx.x & 31;
  const int n   = blockIdx.x * 8 + wid;
  float* S      = smem + wid * 288;  // 16x16 scores
  float* rowsum = S + 256;           // 16
  float* wv     = S + 272;           // 16

  v8f acc = {};
#pragma unroll
  for (int kt = 0; kt < 8; ++kt) {
    v16h a = load_frag_rm(q16, 256, n * 16, kt * 32);
    v16h b = load_frag_rm(v16, 256, n * 16, kt * 32);
    acc = wmma16(a, b, acc);
  }
  {
    const int col = l & 15;
    const int rb  = (l >> 4) << 3;
#pragma unroll
    for (int i = 0; i < 8; ++i) S[(rb + i) * 16 + col] = acc[i];
  }
  __syncthreads();
  if (l < 16) {                       // softmax over m for row l
    float mx = -1e30f;
    for (int m = 0; m < 16; ++m) mx = fmaxf(mx, S[l * 16 + m]);
    float s = 0.f;
    for (int m = 0; m < 16; ++m) { const float e = expf(S[l * 16 + m] - mx); S[l * 16 + m] = e; s += e; }
    rowsum[l] = s;
  }
  __syncthreads();
  if (l < 16) {                       // w[m] = sum_l attn[l][m]
    float w = 0.f;
    for (int r = 0; r < 16; ++r) w += S[r * 16 + l] / rowsum[r];
    wv[l] = w;
  }
  __syncthreads();
  {                                    // ctx[n] = sum_m w[m] * hv[n,m,:]
    const int d0 = l * 8;
    float c[8];
#pragma unroll
    for (int j = 0; j < 8; ++j) c[j] = 0.f;
    for (int m = 0; m < 16; ++m) {
      const float w = wv[m];
      const _Float16* hp = hv + ((size_t)n * 16 + m) * 256 + d0;
#pragma unroll
      for (int j = 0; j < 8; ++j) c[j] += w * (float)hp[j];
    }
#pragma unroll
    for (int j = 0; j < 8; ++j) ctx16[(size_t)n * 256 + d0 + j] = (_Float16)c[j];
  }
}

// ---- broadcast measure context into stacked[:,300:556] ---------------------
__global__ void pack_stacked_ctx(const _Float16* __restrict__ ctx16,
                                 _Float16* __restrict__ stacked) {
  const int idx = blockIdx.x * blockDim.x + threadIdx.x;   // 65536*256
  const int row = idx >> 8, c = idx & 255;
  const int t = row >> 6, b = row & 63;
  const int n = b * 64 + (t >> 4);
  stacked[(size_t)row * 576 + 300 + c] = ctx16[(size_t)n * 256 + c];
}

// ---- cross-entropy reductions ---------------------------------------------
__global__ void ce_pitch(const float* __restrict__ logits,   // [65536,48]
                         const int* __restrict__ tgt, float* __restrict__ accum) {
  const int row = blockIdx.x * blockDim.x + threadIdx.x;
  float nll = 0.f, cnt = 0.f;
  const int tg = tgt[row];
  if (tg != PAD_P) {
    const float* p = logits + (size_t)row * 48;
    float mx = -1e30f;
    for (int j = 0; j < NPITCH; ++j) mx = fmaxf(mx, p[j]);
    float s = 0.f;
    for (int j = 0; j < NPITCH; ++j) s += expf(p[j] - mx);
    nll = mx + logf(s) - p[tg];
    cnt = 1.f;
  }
  for (int o = 16; o > 0; o >>= 1) { nll += __shfl_down(nll, o); cnt += __shfl_down(cnt, o); }
  if ((threadIdx.x & 31) == 0) { atomicAdd(&accum[0], nll); atomicAdd(&accum[1], cnt); }
}

__global__ void ce_ks(const float* __restrict__ logits,      // [4096,16]
                      const int* __restrict__ tgt, float* __restrict__ accum) {
  const int row = blockIdx.x * blockDim.x + threadIdx.x;     // (t,b) flat
  const int t = row >> 6, b = row & 63;
  const int n = b * 64 + (t >> 4);
  float nll = 0.f, cnt = 0.f;
  const int tg = tgt[row];
  if (tg != PAD_K) {
    const float* p = logits + (size_t)n * 16;
    float mx = -1e30f;
    for (int j = 0; j < NKS; ++j) mx = fmaxf(mx, p[j]);
    float s = 0.f;
    for (int j = 0; j < NKS; ++j) s += expf(p[j] - mx);
    nll = mx + logf(s) - p[tg];
    cnt = 1.f;
  }
  for (int o = 16; o > 0; o >>= 1) { nll += __shfl_down(nll, o); cnt += __shfl_down(cnt, o); }
  if ((threadIdx.x & 31) == 0) { atomicAdd(&accum[2], nll); atomicAdd(&accum[3], cnt); }
}

__global__ void finalize(const float* __restrict__ accum, float* __restrict__ out) {
  out[0] = accum[0] / fmaxf(accum[1], 1.f) + accum[2] / fmaxf(accum[3], 1.f);
}

// ---------------------------------------------------------------------------
extern "C" void kernel_launch(void* const* d_in, const int* in_sizes, int n_in,
                              void* d_out, int out_size, void* d_ws, size_t ws_size,
                              hipStream_t stream) {
  (void)in_sizes; (void)n_in; (void)out_size; (void)ws_size;
  const float* sentences = (const float*)d_in[0];
  const int*   pitches   = (const int*)d_in[1];
  const int*   keysigs   = (const int*)d_in[2];
  const float* mWih_f = (const float*)d_in[5],  *mWhh_f = (const float*)d_in[6];
  const float* mbih_f = (const float*)d_in[7],  *mbhh_f = (const float*)d_in[8];
  const float* mWih_b = (const float*)d_in[9],  *mWhh_b = (const float*)d_in[10];
  const float* mbih_b = (const float*)d_in[11], *mbhh_b = (const float*)d_in[12];
  const float* hWih_f = (const float*)d_in[13], *hWhh_f = (const float*)d_in[14];
  const float* hbih_f = (const float*)d_in[15], *hbhh_f = (const float*)d_in[16];
  const float* hWih_b = (const float*)d_in[17], *hWhh_b = (const float*)d_in[18];
  const float* hbih_b = (const float*)d_in[19], *hbhh_b = (const float*)d_in[20];
  const float* Wq = (const float*)d_in[21], *bq = (const float*)d_in[22];
  const float* Wv = (const float*)d_in[23], *bv = (const float*)d_in[24];
  const float* Wp = (const float*)d_in[25], *bp = (const float*)d_in[26];
  const float* Wk = (const float*)d_in[27], *bk = (const float*)d_in[28];

  size_t off = 0;
  auto alloc = [&](size_t bytes) -> void* {
    off = (off + 255) & ~(size_t)255;
    void* p = (char*)d_ws + off;
    off += bytes;
    return p;
  };
  _Float16* x16      = (_Float16*)alloc((size_t)ROWS_TB * 32 * 2);
  _Float16* mWih16f  = (_Float16*)alloc((size_t)464 * 32 * 2);
  _Float16* mWih16b  = (_Float16*)alloc((size_t)464 * 32 * 2);
  _Float16* mWhh16f  = (_Float16*)alloc((size_t)464 * 160 * 2);
  _Float16* mWhh16b  = (_Float16*)alloc((size_t)464 * 160 * 2);
  _Float16* hWih16f  = (_Float16*)alloc((size_t)384 * 320 * 2);
  _Float16* hWih16b  = (_Float16*)alloc((size_t)384 * 320 * 2);
  _Float16* hWhh16f  = (_Float16*)alloc((size_t)384 * 128 * 2);
  _Float16* hWhh16b  = (_Float16*)alloc((size_t)384 * 128 * 2);
  _Float16* Wq16     = (_Float16*)alloc((size_t)256 * 256 * 2);
  _Float16* Wv16     = (_Float16*)alloc((size_t)256 * 256 * 2);
  _Float16* Wp16     = (_Float16*)alloc((size_t)48 * 576 * 2);
  _Float16* Wk16     = (_Float16*)alloc((size_t)16 * 256 * 2);
  _Float16* gi_f     = (_Float16*)alloc((size_t)ROWS_TB * GM * 2);
  _Float16* gi_b     = (_Float16*)alloc((size_t)ROWS_TB * GM * 2);
  _Float16* hx16     = (_Float16*)alloc((size_t)ROWS_TB * 320 * 2);
  _Float16* gih_f    = (_Float16*)alloc((size_t)ROWS_TB * GHI * 2);
  _Float16* gih_b    = (_Float16*)alloc((size_t)ROWS_TB * GHI * 2);
  _Float16* hv16     = (_Float16*)alloc((size_t)ROWS_TB * 256 * 2);
  _Float16* q16      = (_Float16*)alloc((size_t)ROWS_TB * 256 * 2);
  _Float16* v16      = (_Float16*)alloc((size_t)ROWS_TB * 256 * 2);
  _Float16* ctx16    = (_Float16*)alloc((size_t)NSEQ * 256 * 2);
  _Float16* stacked  = (_Float16*)alloc((size_t)ROWS_TB * 576 * 2);
  float*    plog     = (float*)alloc((size_t)ROWS_TB * 48 * 4);
  float*    kslog    = (float*)alloc((size_t)NSEQ * 16 * 4);
  float*    accum    = (float*)alloc(4 * 4);

  auto packN = [&](const float* W, _Float16* o, int N, int K, int Np, int Kp) {
    const int total = Np * Kp;
    pack_w16<<<(total + 255) / 256, 256, 0, stream>>>(W, o, N, K, Np, Kp);
  };

  // zero-init buffers whose padding is read later
  zero_h16<<<2048, 256, 0, stream>>>(hx16, (size_t)ROWS_TB * 320);
  zero_h16<<<2048, 256, 0, stream>>>(stacked, (size_t)ROWS_TB * 576);
  zero_f32<<<1, 32, 0, stream>>>(accum, 4);

  // pack inputs + weights to f16 (row-major [Npad, Kpad])
  packN(sentences, x16, ROWS_TB, D_IN, ROWS_TB, 32);
  packN(mWih_f, mWih16f, GM, D_IN, 464, 32);
  packN(mWih_b, mWih16b, GM, D_IN, 464, 32);
  packN(mWhh_f, mWhh16f, GM, HM, 464, 160);
  packN(mWhh_b, mWhh16b, GM, HM, 464, 160);
  packN(hWih_f, hWih16f, GHI, 300, 384, 320);
  packN(hWih_b, hWih16b, GHI, 300, 384, 320);
  packN(hWhh_f, hWhh16f, GHI, HHI, 384, 128);
  packN(hWhh_b, hWhh16b, GHI, HHI, 384, 128);
  packN(Wq, Wq16, 256, 256, 256, 256);
  packN(Wv, Wv16, 256, 256, 256, 256);
  packN(Wp, Wp16, NPITCH, 556, 48, 576);
  packN(Wk, Wk16, NKS, 256, 16, 256);

  dim3 blk(128);
  // main GRU input gates: [65536,32] x [450,32]^T
  gemm_wmma<<<dim3(ROWS_TB / 64, 29), blk, 0, stream>>>(x16, 32, mWih16f, 32, mbih_f,
                                                        nullptr, 0, gi_f, GM, GM, 1);
  gemm_wmma<<<dim3(ROWS_TB / 64, 29), blk, 0, stream>>>(x16, 32, mWih16b, 32, mbih_b,
                                                        nullptr, 0, gi_b, GM, GM, 1);
  // main GRU scan (persistent, LDS-resident h)
  {
    const size_t sh = (size_t)32 * HMP * 2 + (size_t)32 * HM * 4 + (size_t)32 * 464 * 4;
    main_scan<<<4, 256, sh, stream>>>(gi_f, gi_b, mWhh16f, mWhh16b, mbhh_f, mbhh_b,
                                      hx16, stacked);
  }
  // hier input gates: [65536,320] x [384,320]^T
  gemm_wmma<<<dim3(ROWS_TB / 64, 24), blk, 0, stream>>>(hx16, 320, hWih16f, 320, hbih_f,
                                                        nullptr, 0, gih_f, GHI, GHI, 10);
  gemm_wmma<<<dim3(ROWS_TB / 64, 24), blk, 0, stream>>>(hx16, 320, hWih16b, 320, hbih_b,
                                                        nullptr, 0, gih_b, GHI, GHI, 10);
  // hier GRU scan
  {
    const size_t sh = (size_t)32 * HHI * 2 + (size_t)32 * HHI * 4 + (size_t)32 * GHI * 4;
    hier_scan<<<256, 256, sh, stream>>>(gih_f, gih_b, hWhh16f, hWhh16b, hbhh_f, hbhh_b,
                                        hv16);
  }
  // q/v projections
  gemm_wmma<<<dim3(ROWS_TB / 64, 16), blk, 0, stream>>>(hv16, 256, Wq16, 256, bq,
                                                        nullptr, 0, q16, 256, 256, 8);
  gemm_wmma<<<dim3(ROWS_TB / 64, 16), blk, 0, stream>>>(hv16, 256, Wv16, 256, bv,
                                                        nullptr, 0, v16, 256, 256, 8);
  // attention + context
  attention<<<NSEQ / 8, 256, 0, stream>>>(q16, v16, hv16, ctx16);
  // key-signature head: [4096,256] x [15,256]^T
  gemm_wmma<<<dim3(NSEQ / 64, 1), blk, 0, stream>>>(ctx16, 256, Wk16, 256, bk,
                                                    kslog, 16, nullptr, 0, NKS, 8);
  // broadcast ctx into stacked, then pitch head: [65536,576] x [35,576]^T
  pack_stacked_ctx<<<ROWS_TB, 256, 0, stream>>>(ctx16, stacked);
  gemm_wmma<<<dim3(ROWS_TB / 64, 3), blk, 0, stream>>>(stacked, 576, Wp16, 576, bp,
                                                       plog, 48, nullptr, 0, NPITCH, 18);
  // cross-entropy + finalize
  ce_pitch<<<ROWS_TB / 256, 256, 0, stream>>>(plog, pitches, accum);
  ce_ks<<<ROWS_TB / 256, 256, 0, stream>>>(kslog, keysigs, accum);
  finalize<<<1, 1, 0, stream>>>(accum, (float*)d_out);
}
